// MidAttnBlock_49194555408571
// MI455X (gfx1250) — matrix operand order repaired and checked by reading, compiled, not run on hardware
//
#include <hip/hip_runtime.h>
#include <hip/hip_bf16.h>

typedef __bf16 bf16_t;
typedef __attribute__((ext_vector_type(16))) __bf16 v16bf;
typedef __attribute__((ext_vector_type(8)))  __bf16 v8bf;
typedef __attribute__((ext_vector_type(8)))  float  v8f;
typedef __attribute__((ext_vector_type(4)))  int    v4i;

#define LSEQ 2048
#define CCH  256
#define BB   16
#define C3T  768

// --- CDNA5 async global->LDS copy (ASYNCcnt-tracked), with sync fallback ---
#if defined(__has_builtin)
#if __has_builtin(__builtin_amdgcn_global_load_async_to_lds_b128)
#define USE_ASYNC_LDS 1
#endif
#endif
#ifndef USE_ASYNC_LDS
#define USE_ASYNC_LDS 0
#endif

static __device__ __forceinline__ void copy16_g2l(const bf16_t* g, bf16_t* l) {
#if USE_ASYNC_LDS
  __builtin_amdgcn_global_load_async_to_lds_b128(
      (__attribute__((address_space(1))) v4i*)(unsigned long long)g,
      (__attribute__((address_space(3))) v4i*)l, 0, 0);
#else
  *(uint4*)l = *(const uint4*)g;
#endif
}

static __device__ __forceinline__ void async_wait_all() {
#if USE_ASYNC_LDS
  asm volatile("s_wait_asynccnt 0x0" ::: "memory");
#endif
}

static __device__ __forceinline__ v8f wmma_bf16(v16bf a, v16bf b, v8f c) {
  // D = A(16x32 bf16) * B(32x16 bf16) + C(16x16 f32)
  return __builtin_amdgcn_wmma_f32_16x16x32_bf16(false, a, false, b, (short)0, c,
                                                 false, false);
}

static __device__ __forceinline__ v16bf cat8(v8bf lo, v8bf hi) {
  return __builtin_shufflevector(lo, hi, 0, 1, 2, 3, 4, 5, 6, 7, 8, 9, 10, 11,
                                 12, 13, 14, 15);
}

// ---------------------------------------------------------------------------
// GroupNorm(32 groups) + ReLU, fp32 in -> bf16 out. One block per (b, group).
// ---------------------------------------------------------------------------
__global__ __launch_bounds__(256) void gn_relu_kernel(
    const float* __restrict__ x, const float* __restrict__ gamma,
    const float* __restrict__ beta, bf16_t* __restrict__ out) {
  const int b = blockIdx.x >> 5;
  const int g = blockIdx.x & 31;
  const int tid = threadIdx.x;
  const float* xp = x + ((size_t)b * CCH + (size_t)g * 8) * LSEQ;
  __shared__ float ssum[256];
  __shared__ float ssq[256];
  float s = 0.0f, q = 0.0f;
  for (int i = tid; i < 8 * LSEQ; i += 256) {
    float v = xp[i];
    s += v;
    q += v * v;
  }
  ssum[tid] = s;
  ssq[tid] = q;
  __syncthreads();
  for (int st = 128; st > 0; st >>= 1) {
    if (tid < st) {
      ssum[tid] += ssum[tid + st];
      ssq[tid] += ssq[tid + st];
    }
    __syncthreads();
  }
  const float inv_n = 1.0f / (8.0f * LSEQ);
  const float mean = ssum[0] * inv_n;
  const float var = ssq[0] * inv_n - mean * mean;
  const float inv = rsqrtf(var + 1e-5f);
  bf16_t* op = out + ((size_t)b * CCH + (size_t)g * 8) * LSEQ;
  for (int i = tid; i < 8 * LSEQ; i += 256) {
    int c = g * 8 + (i >> 11);
    float v = (xp[i] - mean) * inv * gamma[c] + beta[c];
    op[i] = (bf16_t)fmaxf(v, 0.0f);
  }
}

// ---------------------------------------------------------------------------
// conv1d (k=taps, pad=taps/2) as WMMA GEMM, double-buffered async LDS staging.
// Block: 256 thr (8 waves) covers 16 o x 128 l.
// ---------------------------------------------------------------------------
__global__ __launch_bounds__(256) void conv_wmma_kernel(
    const bf16_t* __restrict__ hin, const bf16_t* __restrict__ wb,
    const float* __restrict__ bias, const float* __restrict__ tadd,
    const float* __restrict__ resid, float* __restrict__ out_f32,
    bf16_t* __restrict__ out_b16, int OC, int IC, int taps) {
  const int tid = threadIdx.x;
  const int lane = tid & 31;
  const int wv = tid >> 5;
  const int l0b = blockIdx.x * 128;
  const int o0 = blockIdx.y * 16;
  const int b = blockIdx.z;
  const int n16 = lane & 15;
  const int aoff = (lane < 16) ? 0 : 8;   // A K-offset & D row-offset
  const int boff = (lane < 16) ? 0 : 16;  // unused; B uses aoff too
  const bf16_t* hb = hin + (size_t)b * (size_t)IC * LSEQ;

  // cols: 7 = left halo (l0b-1), 8..135 = main (l0b..l0b+127), 136 = right halo
  __shared__ __align__(16) bf16_t in_t[2][32][144];
  __shared__ __align__(16) bf16_t wt_t[2][3][16][32];

  auto stage = [&](int buf, int kc) {
    // main 32x128 input block: 512 b128 chunks of 8 bf16
    for (int ch = tid; ch < 512; ch += 256) {
      const int row = ch >> 4, cc = ch & 15;
      copy16_g2l(hb + (size_t)(kc + row) * LSEQ + l0b + cc * 8,
                 &in_t[buf][row][8 + cc * 8]);
    }
    // halo columns (scalar, sync)
    if (tid < 32) {
      const int gl = l0b - 1;
      in_t[buf][tid][7] =
          (gl >= 0) ? hb[(size_t)(kc + tid) * LSEQ + gl] : (bf16_t)0.0f;
    } else if (tid < 64) {
      const int row = tid - 32;
      const int gl = l0b + 128;
      in_t[buf][row][136] =
          (gl < LSEQ) ? hb[(size_t)(kc + row) * LSEQ + gl] : (bf16_t)0.0f;
    }
    // weight tile: taps x 16 o x 32 i -> taps*64 b128 chunks
    for (int ch = tid; ch < taps * 64; ch += 256) {
      const int tap = ch >> 6, rem = ch & 63;
      const int orow = rem >> 2, cc = rem & 3;
      copy16_g2l(wb + (size_t)tap * OC * IC + (size_t)(o0 + orow) * IC + kc +
                     cc * 8,
                 &wt_t[buf][tap][orow][cc * 8]);
    }
  };

  v8f acc = {};
  stage(0, 0);
  async_wait_all();
  __syncthreads();

  for (int kci = 0; kci < IC / 32; ++kci) {
    const int kc = kci * 32;
    const int cur = kci & 1;
    if (kc + 32 < IC) stage(cur ^ 1, kc + 32);

    for (int tap = 0; tap < taps; ++tap) {
      const int shift = (taps == 3) ? (tap - 1) : 0;
      // A fragment: two contiguous v8bf runs per lane
      v8bf alo = *(const v8bf*)(&wt_t[cur][tap][n16][aoff]);
      v8bf ahi = *(const v8bf*)(&wt_t[cur][tap][n16][aoff + 16]);
      v16bf a = cat8(alo, ahi);
      // B fragment: column read (strided over channel rows)
      const int colL = 8 + wv * 16 + n16 + shift;
      v16bf bm;
#pragma unroll
      for (int e = 0; e < 16; ++e) bm[e] = in_t[cur][((e < 8) ? e : e + 8) + aoff][colL];
      acc = wmma_bf16(a, bm, acc);
    }
    async_wait_all();
    __syncthreads();
  }

  const int l = l0b + wv * 16 + n16;
#pragma unroll
  for (int r = 0; r < 8; ++r) {
    const int o = o0 + r + aoff;
    float v = acc[r] + bias[o];
    if (tadd) v += tadd[b * OC + o];
    const size_t idx = ((size_t)b * OC + o) * LSEQ + l;
    if (resid) v += resid[idx];
    if (out_f32) out_f32[idx] = v;
    if (out_b16) out_b16[idx] = (bf16_t)v;
  }
}

// ---------------------------------------------------------------------------
// Flash attention, double-buffered async-staged K/V. kqv: bf16 (B,768,L).
// Block: 128 thr (4 waves) share the kl stream; K/V tiles (256x32 bf16) are
// async-loaded into the alternate buffer while the current one is consumed.
// One wave = one 16-row Q tile, full 16x256 O tile in registers.
// ---------------------------------------------------------------------------
__global__ __launch_bounds__(128) void flash_attn_kernel(
    const bf16_t* __restrict__ kqv, float* __restrict__ av) {
  const int tid = threadIdx.x;
  const int lane = tid & 31;
  const int wv = tid >> 5;
  const int b = blockIdx.x >> 5;
  const int qg = blockIdx.x & 31;
  const int q0 = (qg * 4 + wv) * 16;

  const bf16_t* kmat = kqv + (size_t)b * C3T * LSEQ;
  const bf16_t* qmat = kmat + (size_t)CCH * LSEQ;
  const bf16_t* vmat = kmat + (size_t)2 * CCH * LSEQ;

  __shared__ __align__(16) bf16_t kt[2][256][32];
  __shared__ __align__(16) bf16_t vt[2][256][32];
  __shared__ __align__(16) bf16_t plds[4][16][32];

  const int n16 = lane & 15;
  const int aoff = (lane < 16) ? 0 : 8;
  const int boff = (lane < 16) ? 0 : 16;

  auto stage = [&](int buf, int kl0) {
    for (int ch = tid; ch < 1024; ch += 128) {
      const int row = ch >> 2, cc = ch & 3;
      const size_t goff = (size_t)row * LSEQ + kl0 + cc * 8;
      copy16_g2l(kmat + goff, &kt[buf][row][cc * 8]);
      copy16_g2l(vmat + goff, &vt[buf][row][cc * 8]);
      if (kl0 + 32 < LSEQ) {  // prime L2 one tile beyond the double buffer
        __builtin_prefetch((const void*)(kmat + goff + 32), 0, 1);
        __builtin_prefetch((const void*)(vmat + goff + 32), 0, 1);
      }
    }
  };

  // Q^T A-fragments (M = query pos, K = channel), loaded once
  v16bf aq[8];
#pragma unroll
  for (int kc = 0; kc < 8; ++kc) {
#pragma unroll
    for (int e = 0; e < 16; ++e) {
      int c = kc * 32 + ((e < 8) ? e : e + 8) + aoff;
      aq[kc][e] = qmat[(size_t)c * LSEQ + q0 + n16];
    }
  }

  const v8f vzero = {};
  v8f oacc[16];
#pragma unroll
  for (int t = 0; t < 16; ++t) oacc[t] = vzero;
  float mi[8], li[8];
#pragma unroll
  for (int r = 0; r < 8; ++r) {
    mi[r] = -1e30f;
    li[r] = 0.0f;
  }

  const float scale = 0.0625f;  // 1/sqrt(256)

  stage(0, 0);
  async_wait_all();
  __syncthreads();

  for (int it = 0; it < LSEQ / 32; ++it) {
    const int kl0 = it * 32;
    const int cur = it & 1;
    if (kl0 + 32 < LSEQ) stage(cur ^ 1, kl0 + 32);

    // S tiles: S[ql, kl] = sum_c q[c,ql] k[c,kl]
    v8f s0 = vzero, s1 = vzero;
    for (int kc = 0; kc < 8; ++kc) {
      v16bf bk0, bk1;
#pragma unroll
      for (int e = 0; e < 16; ++e) {
        bk0[e] = kt[cur][kc * 32 + ((e < 8) ? e : e + 8) + aoff][n16];
        bk1[e] = kt[cur][kc * 32 + ((e < 8) ? e : e + 8) + aoff][n16 + 16];
      }
      s0 = wmma_bf16(aq[kc], bk0, s0);
      s1 = wmma_bf16(aq[kc], bk1, s1);
    }

    // Online softmax per row (rows split across the two 16-lane halves)
#pragma unroll
    for (int r = 0; r < 8; ++r) {
      float v0 = s0[r] * scale;
      float v1 = s1[r] * scale;
      float mx = fmaxf(v0, v1);
#pragma unroll
      for (int m = 1; m < 16; m <<= 1) mx = fmaxf(mx, __shfl_xor(mx, m, 32));
      const float mnew = fmaxf(mi[r], mx);
      const float p0 = __expf(v0 - mnew);
      const float p1 = __expf(v1 - mnew);
      const float alpha = __expf(mi[r] - mnew);
      float rs = p0 + p1;
#pragma unroll
      for (int m = 1; m < 16; m <<= 1) rs += __shfl_xor(rs, m, 32);
      li[r] = li[r] * alpha + rs;
      mi[r] = mnew;
#pragma unroll
      for (int t = 0; t < 16; ++t) oacc[t][r] *= alpha;
      plds[wv][r + aoff][n16] = (bf16_t)p0;
      plds[wv][r + aoff][n16 + 16] = (bf16_t)p1;
    }
    // Per-wave LDS ops are in-order; wait + compiler barrier before re-read.
    asm volatile("s_wait_dscnt 0" ::: "memory");
    // P re-layout D->A: two contiguous v8bf runs per lane
    v8bf plo = *(const v8bf*)(&plds[wv][n16][aoff]);
    v8bf phi = *(const v8bf*)(&plds[wv][n16][aoff + 16]);
    v16bf ap = cat8(plo, phi);

    // O[ql, c] += P[ql, kl] * V^T[kl, c]; V fragment rows are contiguous
#pragma unroll
    for (int t = 0; t < 16; ++t) {
      v8bf vlo = *(const v8bf*)(&vt[cur][t * 16 + n16][aoff]);
      v8bf vhi = *(const v8bf*)(&vt[cur][t * 16 + n16][aoff + 16]);
      oacc[t] = wmma_bf16(ap, cat8(vlo, vhi), oacc[t]);
    }

    async_wait_all();
    __syncthreads();
  }

  float* avb = av + (size_t)b * CCH * LSEQ;
#pragma unroll
  for (int t = 0; t < 16; ++t) {
#pragma unroll
    for (int r = 0; r < 8; ++r) {
      const int c = t * 16 + n16;
      const int ql = q0 + r + aoff;
      avb[(size_t)c * LSEQ + ql] = oacc[t][r] / li[r];
    }
  }
}

// ---------------------------------------------------------------------------
// Weight conversion: f32 (OC, IC, 3) -> bf16 [tap][o][i]; and flat f32->bf16.
// ---------------------------------------------------------------------------
__global__ void conv_w_bf16_kernel(const float* __restrict__ w,
                                   bf16_t* __restrict__ wb, int OC, int IC) {
  const int n = OC * IC * 3;
  for (int i = blockIdx.x * blockDim.x + threadIdx.x; i < n;
       i += gridDim.x * blockDim.x) {
    const int tap = i % 3;
    const int oi = i / 3;
    wb[(size_t)tap * OC * IC + oi] = (bf16_t)w[i];
  }
}

__global__ void flat_w_bf16_kernel(const float* __restrict__ w,
                                   bf16_t* __restrict__ wb, int n) {
  for (int i = blockIdx.x * blockDim.x + threadIdx.x; i < n;
       i += gridDim.x * blockDim.x)
    wb[i] = (bf16_t)w[i];
}

// ---------------------------------------------------------------------------
extern "C" void kernel_launch(void* const* d_in, const int* in_sizes, int n_in,
                              void* d_out, int out_size, void* d_ws,
                              size_t ws_size, hipStream_t stream) {
  (void)in_sizes; (void)n_in; (void)out_size; (void)ws_size;
  const float* x        = (const float*)d_in[0];
  const float* t        = (const float*)d_in[1];
  const float* r1_gn1_w = (const float*)d_in[2];
  const float* r1_gn1_b = (const float*)d_in[3];
  const float* r1_c1_w  = (const float*)d_in[4];
  const float* r1_c1_b  = (const float*)d_in[5];
  const float* r1_gn2_w = (const float*)d_in[6];
  const float* r1_gn2_b = (const float*)d_in[7];
  const float* r1_c2_w  = (const float*)d_in[8];
  const float* r1_c2_b  = (const float*)d_in[9];
  const float* r2_gn1_w = (const float*)d_in[10];
  const float* r2_gn1_b = (const float*)d_in[11];
  const float* r2_c1_w  = (const float*)d_in[12];
  const float* r2_c1_b  = (const float*)d_in[13];
  const float* r2_gn2_w = (const float*)d_in[14];
  const float* r2_gn2_b = (const float*)d_in[15];
  const float* r2_c2_w  = (const float*)d_in[16];
  const float* r2_c2_b  = (const float*)d_in[17];
  const float* lin_w    = (const float*)d_in[18];
  const float* lin_b    = (const float*)d_in[19];

  char* ws = (char*)d_ws;
  const size_t NCL = (size_t)BB * CCH * LSEQ;  // 8,388,608 elements
  bf16_t* hb   = (bf16_t*)(ws);                // NCL bf16
  float*  y1   = (float*)(ws + NCL * 2);       // NCL f32
  bf16_t* x1b  = (bf16_t*)(ws + NCL * 6);      // NCL bf16
  bf16_t* kqvb = (bf16_t*)(ws + NCL * 8);      // 3*NCL bf16
  float*  av   = (float*)(ws + NCL * 14);      // NCL f32
  bf16_t* w1   = (bf16_t*)(ws + NCL * 18);     // 3*256*256 each
  bf16_t* w2   = w1 + 196608;
  bf16_t* w3   = w2 + 196608;
  bf16_t* w4   = w3 + 196608;
  bf16_t* wl   = w4 + 196608;                  // 768*256

  conv_w_bf16_kernel<<<192, 256, 0, stream>>>(r1_c1_w, w1, 256, 256);
  conv_w_bf16_kernel<<<192, 256, 0, stream>>>(r1_c2_w, w2, 256, 256);
  conv_w_bf16_kernel<<<192, 256, 0, stream>>>(r2_c1_w, w3, 256, 256);
  conv_w_bf16_kernel<<<192, 256, 0, stream>>>(r2_c2_w, w4, 256, 256);
  flat_w_bf16_kernel<<<192, 256, 0, stream>>>(lin_w, wl, C3T * CCH);

  const dim3 cgrid(16, 16, 16);  // l-blocks of 128, o-tiles, batch
  const dim3 kgrid(16, 48, 16);  // kqv: OC = 768

  // --- ResBlock 1 ---
  gn_relu_kernel<<<512, 256, 0, stream>>>(x, r1_gn1_w, r1_gn1_b, hb);
  conv_wmma_kernel<<<cgrid, 256, 0, stream>>>(hb, w1, r1_c1_b, t, nullptr, y1,
                                              nullptr, 256, 256, 3);
  gn_relu_kernel<<<512, 256, 0, stream>>>(y1, r1_gn2_w, r1_gn2_b, hb);
  conv_wmma_kernel<<<cgrid, 256, 0, stream>>>(hb, w2, r1_c2_b, nullptr, x,
                                              nullptr, x1b, 256, 256, 3);

  // --- kqv 1x1 conv ---
  conv_wmma_kernel<<<kgrid, 256, 0, stream>>>(x1b, wl, lin_b, nullptr, nullptr,
                                              nullptr, kqvb, 768, 256, 1);

  // --- Flash attention ---
  flash_attn_kernel<<<512, 128, 0, stream>>>(kqvb, av);

  // --- ResBlock 2 ---
  gn_relu_kernel<<<512, 256, 0, stream>>>(av, r2_gn1_w, r2_gn1_b, hb);
  conv_wmma_kernel<<<cgrid, 256, 0, stream>>>(hb, w3, r2_c1_b, t, nullptr, y1,
                                              nullptr, 256, 256, 3);
  gn_relu_kernel<<<512, 256, 0, stream>>>(y1, r2_gn2_w, r2_gn2_b, hb);
  conv_wmma_kernel<<<cgrid, 256, 0, stream>>>(hb, w4, r2_c2_b, nullptr, av,
                                              (float*)d_out, nullptr, 256, 256,
                                              3);
}
